// SelectiveScan1D_6579889897505
// MI455X (gfx1250) — matrix operand order repaired and checked by reading, not compiled, not run on hardware
//
#include <hip/hip_runtime.h>
#include <math.h>

// ---- problem constants (match reference) ----
#define DIMC  1024
#define HID   2048
#define MHC   256
#define GRP   4
#define KW    5
#define BATCH 8
#define SEQL  2048

typedef __bf16 bf16;
typedef __attribute__((ext_vector_type(16))) __bf16 v16bf;
typedef __attribute__((ext_vector_type(8)))  float  v8f;

// ---- GEMM tiling: block 64x128, 128 threads (4 waves), each wave 32x64 ----
#define BM 64
#define BN 128
#define BK 32
#define LDSS 40   // bf16 elems/row: 32 + 8 pad -> 80B row stride (8B aligned b64 stores)

// ---- epilogue modes ----
#define EP_NONE      0
#define EP_BIAS      1
#define EP_BIAS_SILU 2   // bias; silu where (g*M + row) < silu_limit  (w_in: u half only)
#define EP_A_GATE    3   // DMIN + (DMAX-DMIN)*sigmoid(v + bias)
#define EP_TANH      4   // tanh(v + bias)
#define EP_RESID     5   // v += R[row, col]

union Pk4 { bf16 h[4]; uint2 u; };

// A fragment: 16x32 bf16 (M x K). lanes 0-15 = rows, half=lane>>4 selects K-octets.
__device__ __forceinline__ v16bf load_a_frag(const bf16* As, int m_off, int lane) {
  const int row  = m_off + (lane & 15);
  const int half = lane >> 4;
  const bf16* p = As + row * LDSS;
  union { unsigned u[8]; v16bf v; } f;
#pragma unroll
  for (int r = 0; r < 8; ++r) {
    const int kk = (r < 4) ? (8 * half + 2 * r) : (16 + 8 * half + 2 * (r - 4));
    f.u[r] = *(const unsigned*)(p + kk);       // two bf16 (K pair) per dword
  }
  return f.v;
}

// B fragment: 32x16 bf16 (K x N), Bs stored n-major [n][k] so K pairs are contiguous.
__device__ __forceinline__ v16bf load_b_frag(const bf16* Bs, int n_off, int lane) {
  const int col  = n_off + (lane & 15);
  const int half = lane >> 4;
  const bf16* p = Bs + col * LDSS;
  union { unsigned u[8]; v16bf v; } f;
#pragma unroll
  for (int r = 0; r < 8; ++r) f.u[r] = *(const unsigned*)(p + ((r < 4) ? (8 * half + 2 * r) : (16 + 8 * half + 2 * (r - 4))));
  return f.v;
}

// Generic (grouped) 1x1-conv GEMM:  Y[b, g*M+m, n] = epilogue( sum_k W[g*M+m, k] * X[b, g*K+k, n] )
// Double-buffered LDS: one barrier per K-step; global loads pipelined 2 steps ahead.
__global__ __launch_bounds__(128)
void gemm_bf16_kernel(const float* __restrict__ X,
                      const float* __restrict__ W,
                      const float* __restrict__ bias,
                      const float* __restrict__ R,
                      float* __restrict__ Y,
                      int M, int N, int K, int G,
                      long xBatchStride, long yBatchStride,
                      int ep, int silu_limit)
{
  __shared__ __align__(16) bf16 As[2][BM * LDSS];
  __shared__ __align__(16) bf16 Bs[2][BN * LDSS];

  const int tid  = threadIdx.x;
  const int lane = tid & 31;
  const int wave = tid >> 5;
  const int wm   = (wave >> 1) * 32;   // 2 waves along M
  const int wn   = (wave & 1)  * 64;   // 2 waves along N

  const int bz = blockIdx.z;
  const int b  = bz / G;
  const int g  = bz % G;

  const float* Xb = X + (long)b * xBatchStride + (long)g * K * N;
  const float* Wb = W + (long)g * M * K;
  const float* Bb = bias ? (bias + (long)g * M) : nullptr;
  const float* Rb = R ? (R + (long)b * yBatchStride + (long)g * M * N) : nullptr;
  float*       Yb = Y + (long)b * yBatchStride + (long)g * M * N;

  const int m0 = blockIdx.y * BM;
  const int n0 = blockIdx.x * BN;

  // staging index precompute
  const int aRow = tid >> 3;           // +16 per i : 4 float4 / thread, A = 64x32
  const int aKq  = (tid & 7) << 2;
  const int bNq  = (tid & 31) << 2;    // sub-block 4k x 4n; 2 rounds (s = tid + r*128)
  const int bKq0 = (tid >> 5) << 2;    // round r adds 16 to kq

  const v8f vzero = {0.f,0.f,0.f,0.f,0.f,0.f,0.f,0.f};
  v8f acc[2][4];
#pragma unroll
  for (int i = 0; i < 2; ++i)
#pragma unroll
    for (int j = 0; j < 4; ++j) acc[i][j] = vzero;

  float4 aR[4];   // A slab registers (4 rows' quads)
  float4 bR[8];   // B slab registers (2 sub-blocks x 4 k-rows)

  const int nk = K / BK;

  // ---- register load helper (macro-ish lambdas) ----
  auto load_regs = [&](int k0) {
#pragma unroll
    for (int i = 0; i < 4; ++i)
      aR[i] = *(const float4*)&Wb[(long)(m0 + aRow + i * 16) * K + k0 + aKq];
#pragma unroll
    for (int r = 0; r < 2; ++r)
#pragma unroll
      for (int kk = 0; kk < 4; ++kk)
        bR[r * 4 + kk] = *(const float4*)&Xb[(long)(k0 + bKq0 + r * 16 + kk) * N + n0 + bNq];
  };
  auto store_lds = [&](int buf) {
    bf16* Asb = As[buf];
    bf16* Bsb = Bs[buf];
#pragma unroll
    for (int i = 0; i < 4; ++i) {
      Pk4 pk;
      pk.h[0] = (bf16)aR[i].x; pk.h[1] = (bf16)aR[i].y;
      pk.h[2] = (bf16)aR[i].z; pk.h[3] = (bf16)aR[i].w;
      *(uint2*)&Asb[(aRow + i * 16) * LDSS + aKq] = pk.u;
    }
#pragma unroll
    for (int r = 0; r < 2; ++r) {
      float fr[4][4];
#pragma unroll
      for (int kk = 0; kk < 4; ++kk) {
        fr[kk][0] = bR[r * 4 + kk].x; fr[kk][1] = bR[r * 4 + kk].y;
        fr[kk][2] = bR[r * 4 + kk].z; fr[kk][3] = bR[r * 4 + kk].w;
      }
      const int kq = bKq0 + r * 16;
#pragma unroll
      for (int j = 0; j < 4; ++j) {
        Pk4 pk;
#pragma unroll
        for (int kk = 0; kk < 4; ++kk) pk.h[kk] = (bf16)fr[kk][j];
        *(uint2*)&Bsb[(bNq + j) * LDSS + kq] = pk.u;
      }
    }
  };

  // ---- prologue: fill buffer 0, prefetch k-step 1 into registers ----
  load_regs(0);
  store_lds(0);
  if (nk > 1) load_regs(BK);
  __syncthreads();

  for (int kb = 0; kb < nk; ++kb) {
    const int cur = kb & 1;

    // fragments from current buffer
    v16bf af0 = load_a_frag(As[cur], wm + 0,  lane);
    v16bf af1 = load_a_frag(As[cur], wm + 16, lane);
    v16bf bf0 = load_b_frag(Bs[cur], wn + 0,  lane);
    v16bf bf1 = load_b_frag(Bs[cur], wn + 16, lane);
    v16bf bf2 = load_b_frag(Bs[cur], wn + 32, lane);
    v16bf bf3 = load_b_frag(Bs[cur], wn + 48, lane);

    // stage next step into the other buffer; prefetch step kb+2 into registers
    if (kb + 1 < nk) {
      store_lds(cur ^ 1);
      if (kb + 2 < nk) load_regs((kb + 2) * BK);
    }

    acc[0][0] = __builtin_amdgcn_wmma_f32_16x16x32_bf16(false, af0, false, bf0, (short)0, acc[0][0], false, false);
    acc[0][1] = __builtin_amdgcn_wmma_f32_16x16x32_bf16(false, af0, false, bf1, (short)0, acc[0][1], false, false);
    acc[0][2] = __builtin_amdgcn_wmma_f32_16x16x32_bf16(false, af0, false, bf2, (short)0, acc[0][2], false, false);
    acc[0][3] = __builtin_amdgcn_wmma_f32_16x16x32_bf16(false, af0, false, bf3, (short)0, acc[0][3], false, false);
    acc[1][0] = __builtin_amdgcn_wmma_f32_16x16x32_bf16(false, af1, false, bf0, (short)0, acc[1][0], false, false);
    acc[1][1] = __builtin_amdgcn_wmma_f32_16x16x32_bf16(false, af1, false, bf1, (short)0, acc[1][1], false, false);
    acc[1][2] = __builtin_amdgcn_wmma_f32_16x16x32_bf16(false, af1, false, bf2, (short)0, acc[1][2], false, false);
    acc[1][3] = __builtin_amdgcn_wmma_f32_16x16x32_bf16(false, af1, false, bf3, (short)0, acc[1][3], false, false);

    if (kb + 1 < nk) __syncthreads();   // next buffer ready; current buffer fully consumed
  }

  // C layout: vgpr r -> row = 8*half + r ; col = lane&15
  const int half = lane >> 4;
  const int colb = lane & 15;
#pragma unroll
  for (int i = 0; i < 2; ++i) {
#pragma unroll
    for (int j = 0; j < 4; ++j) {
#pragma unroll
      for (int r = 0; r < 8; ++r) {
        const int row = m0 + wm + i * 16 + half * 8 + r;   // row within group [0, M)
        const int col = n0 + wn + j * 16 + colb;
        float v = acc[i][j][r];
        if (ep == EP_BIAS || ep == EP_BIAS_SILU || ep == EP_A_GATE || ep == EP_TANH)
          v += Bb[row];
        if (ep == EP_BIAS_SILU) {
          if (g * M + row < silu_limit) v = v / (1.f + __expf(-v));   // silu
        } else if (ep == EP_A_GATE) {
          v = 0.9f + 0.099f * (1.f / (1.f + __expf(-v)));            // DMIN..DMAX sigmoid
        } else if (ep == EP_TANH) {
          v = tanhf(v);
        } else if (ep == EP_RESID) {
          v += Rb[(long)row * N + col];
        }
        Yb[(long)row * N + col] = v;
      }
    }
  }
}

// depthwise conv K=5, pad=2, over u = first HID channels of t [B, 2*HID, L]
__global__ __launch_bounds__(256)
void dwconv_kernel(const float* __restrict__ T, const float* __restrict__ Wd,
                   float* __restrict__ U)
{
  long idx = (long)blockIdx.x * 256 + threadIdx.x;
  const long total = (long)BATCH * HID * SEQL;
  if (idx >= total) return;
  const int l = (int)(idx % SEQL);
  const long t2 = idx / SEQL;
  const int c = (int)(t2 % HID);
  const int b = (int)(t2 / HID);
  const float* in = T + ((long)b * (2 * HID) + c) * SEQL;
  float acc = 0.f;
#pragma unroll
  for (int k = 0; k < KW; ++k) {
    const int ll = l + k - (KW / 2);
    if (ll >= 0 && ll < SEQL) acc += Wd[c * KW + k] * in[ll];
  }
  U[idx] = acc;
}

// gmix = u2 * sigmoid(gate); vectorized float4 (L multiple of 4)
__global__ __launch_bounds__(256)
void gate_kernel(const float* __restrict__ U2, const float* __restrict__ T,
                 float* __restrict__ Gm)
{
  long idx4 = (long)blockIdx.x * 256 + threadIdx.x;
  const long total4 = (long)BATCH * HID * SEQL / 4;
  if (idx4 >= total4) return;
  const long idx = idx4 * 4;
  const int l = (int)(idx % SEQL);
  const long t2 = idx / SEQL;
  const int c = (int)(t2 % HID);
  const int b = (int)(t2 / HID);
  const float4 u = *(const float4*)&U2[idx];
  const float4 gv = *(const float4*)&T[((long)b * (2 * HID) + HID + c) * SEQL + l];
  float4 o;
  o.x = u.x * (1.f / (1.f + __expf(-gv.x)));
  o.y = u.y * (1.f / (1.f + __expf(-gv.y)));
  o.z = u.z * (1.f / (1.f + __expf(-gv.z)));
  o.w = u.w * (1.f / (1.f + __expf(-gv.w)));
  *(float4*)&Gm[idx] = o;
}

// Parallel linear-recurrence scan per (b, mh) row: h_t = v_t + a_t*h_{t-1}, v = b*s.
// Then y = c*h + d_s*s. Block = 256 threads, chunk = 8 elems/thread, pair-scan in LDS.
__global__ __launch_bounds__(256)
void scan_kernel(const float* __restrict__ S, const float* __restrict__ A,
                 const float* __restrict__ Bv, const float* __restrict__ C,
                 const float* __restrict__ Ds, float* __restrict__ Yout)
{
  const int bc = blockIdx.x;              // b*MH + mh
  const int mh = bc % MHC;
  const long base = (long)bc * SEQL;
  __shared__ float sA[SEQL];
  __shared__ float sV[SEQL];
  __shared__ float cAagg[256];
  __shared__ float cVagg[256];
  const int tid = threadIdx.x;

  for (int l = tid; l < SEQL; l += 256) {
    float a = A[base + l];
    a = fminf(fmaxf(a, 1e-4f), 0.9999f);
    sA[l] = a;
    sV[l] = Bv[base + l] * S[base + l];
  }
  __syncthreads();

  // serial scan within chunk of 8; store local inclusive v and running a-product
  const int c0 = tid * 8;
  float pa = 1.f, pv = 0.f;
#pragma unroll
  for (int i = 0; i < 8; ++i) {
    const float a = sA[c0 + i], v = sV[c0 + i];
    pv = v + a * pv;
    pa = pa * a;
    sA[c0 + i] = pa;
    sV[c0 + i] = pv;
  }
  cAagg[tid] = pa; cVagg[tid] = pv;
  __syncthreads();

  // Hillis-Steele inclusive scan of 256 chunk aggregates: (a,v)2 o (a,v)1 = (a1*a2, v2 + a2*v1)
  for (int off = 1; off < 256; off <<= 1) {
    const float a2 = cAagg[tid], v2 = cVagg[tid];
    float na = a2, nv = v2;
    const int src = tid - off;
    if (src >= 0) { na = cAagg[src] * a2; nv = v2 + a2 * cVagg[src]; }
    __syncthreads();
    cAagg[tid] = na; cVagg[tid] = nv;
    __syncthreads();
  }

  const float ds = Ds[mh];
  for (int l = tid; l < SEQL; l += 256) {
    const int chunk = l >> 3;
    const float hp = (chunk == 0) ? 0.f : cVagg[chunk - 1];  // exclusive prefix state
    const float h = sV[l] + sA[l] * hp;
    Yout[base + l] = C[base + l] * h + ds * S[base + l];
  }
}

extern "C" void kernel_launch(void* const* d_in, const int* in_sizes, int n_in,
                              void* d_out, int out_size, void* d_ws, size_t ws_size,
                              hipStream_t stream)
{
  (void)in_sizes; (void)n_in; (void)out_size; (void)ws_size;
  const float* x     = (const float*)d_in[0];
  const float* w_in  = (const float*)d_in[1];
  const float* b_in  = (const float*)d_in[2];
  const float* w_dw  = (const float*)d_in[3];
  const float* w_si  = (const float*)d_in[4];
  const float* w_sd  = (const float*)d_in[5];
  const float* b_sd  = (const float*)d_in[6];
  const float* w_sb  = (const float*)d_in[7];
  const float* b_sb  = (const float*)d_in[8];
  const float* w_sc  = (const float*)d_in[9];
  const float* b_sc  = (const float*)d_in[10];
  const float* d_s   = (const float*)d_in[11];
  const float* w_so  = (const float*)d_in[12];
  const float* w_mix = (const float*)d_in[13];
  const float* b_mix = (const float*)d_in[14];
  const float* w_out = (const float*)d_in[15];
  const float* b_out = (const float*)d_in[16];
  float* out = (float*)d_out;

  float* ws = (float*)d_ws;
  const size_t t_sz   = (size_t)BATCH * 2 * HID * SEQL;
  const size_t udw_sz = (size_t)BATCH * HID * SEQL;
  const size_t ch_sz  = (size_t)BATCH * MHC * SEQL;
  float* t_buf  = ws;                    // [B,4096,L]; reused as w_mix output later
  float* udw    = t_buf + t_sz;          // [B,2048,L]; reused as gate product later
  float* s_buf  = udw + udw_sz;
  float* a_buf  = s_buf + ch_sz;
  float* bb_buf = a_buf + ch_sz;
  float* c_buf  = bb_buf + ch_sz;
  float* my_buf = c_buf + ch_sz;
  float* u2     = my_buf + ch_sz;        // [B,2048,L]

  const dim3 blk(128);
  const long elw_total = (long)BATCH * HID * SEQL;
  const int  elw_grid  = (int)((elw_total + 255) / 256);
  const int  elw_grid4 = (int)((elw_total / 4 + 255) / 256);

  // 1) t = w_in (grouped) + bias, silu on u half (rows < HID)
  {
    dim3 grid(SEQL / BN, (2 * HID / GRP) / BM, BATCH * GRP);
    gemm_bf16_kernel<<<grid, blk, 0, stream>>>(x, w_in, b_in, nullptr, t_buf,
        2 * HID / GRP, SEQL, DIMC / GRP, GRP,
        (long)DIMC * SEQL, (long)2 * HID * SEQL, EP_BIAS_SILU, HID);
  }
  // 2) depthwise conv -> u_dw
  dwconv_kernel<<<elw_grid, 256, 0, stream>>>(t_buf, w_dw, udw);
  // 3-6) SSM projections
  {
    dim3 grid(SEQL / BN, MHC / BM, BATCH);
    gemm_bf16_kernel<<<grid, blk, 0, stream>>>(udw, w_si, nullptr, nullptr, s_buf,
        MHC, SEQL, HID, 1, (long)HID * SEQL, (long)MHC * SEQL, EP_NONE, 0);
    gemm_bf16_kernel<<<grid, blk, 0, stream>>>(udw, w_sd, b_sd, nullptr, a_buf,
        MHC, SEQL, HID, 1, (long)HID * SEQL, (long)MHC * SEQL, EP_A_GATE, 0);
    gemm_bf16_kernel<<<grid, blk, 0, stream>>>(udw, w_sb, b_sb, nullptr, bb_buf,
        MHC, SEQL, HID, 1, (long)HID * SEQL, (long)MHC * SEQL, EP_TANH, 0);
    gemm_bf16_kernel<<<grid, blk, 0, stream>>>(udw, w_sc, b_sc, nullptr, c_buf,
        MHC, SEQL, HID, 1, (long)HID * SEQL, (long)MHC * SEQL, EP_TANH, 0);
  }
  // 7) selective scan -> mamba_y
  scan_kernel<<<BATCH * MHC, 256, 0, stream>>>(s_buf, a_buf, bb_buf, c_buf, d_s, my_buf);
  // 8) u2 = u_dw + w_so(mamba_y)
  {
    dim3 grid(SEQL / BN, HID / BM, BATCH);
    gemm_bf16_kernel<<<grid, blk, 0, stream>>>(my_buf, w_so, nullptr, udw, u2,
        HID, SEQL, MHC, 1, (long)MHC * SEQL, (long)HID * SEQL, EP_RESID, 0);
  }
  // 9) gmix = u2 * sigmoid(gate)   (overwrites udw; u_dw dead after step 8)
  gate_kernel<<<elw_grid4, 256, 0, stream>>>(u2, t_buf, udw);
  // 10) mix = w_mix(gmix) grouped + bias  (into t_buf region; t dead after step 9)
  {
    dim3 grid(SEQL / BN, (HID / GRP) / BM, BATCH * GRP);
    gemm_bf16_kernel<<<grid, blk, 0, stream>>>(udw, w_mix, b_mix, nullptr, t_buf,
        HID / GRP, SEQL, HID / GRP, GRP,
        (long)HID * SEQL, (long)HID * SEQL, EP_BIAS, 0);
  }
  // 11) out = w_out(mix) + bias
  {
    dim3 grid(SEQL / BN, DIMC / BM, BATCH);
    gemm_bf16_kernel<<<grid, blk, 0, stream>>>(t_buf, w_out, b_out, nullptr, out,
        DIMC, SEQL, HID, 1, (long)HID * SEQL, (long)DIMC * SEQL, EP_BIAS, 0);
  }
}
